// MultiHeadAttention_2568390443457
// MI455X (gfx1250) — matrix operand order repaired and checked by reading, not compiled, not run on hardware
//
#include <hip/hip_runtime.h>
#include <hip/hip_bf16.h>

typedef __bf16 bf16;
typedef __attribute__((ext_vector_type(16))) __bf16 v16bf;
typedef __attribute__((ext_vector_type(8)))  __bf16 v8bf;
typedef __attribute__((ext_vector_type(8)))  float  v8f;

#define NH 16
#define NB 2
#define LL 2048
#define DM 1024
#define DK 64

// workspace layout (bf16 element offsets)
#define OFF_QS   ((size_t)0)         // [H][B][L][64]
#define OFF_KS   ((size_t)4194304)   // [H][B][L][64]
#define OFF_VST  ((size_t)8388608)   // [H][B][64][L]   (transposed v_s)
#define OFF_CAT  ((size_t)12582912)  // [B][L][H*64]
#define OFF_WQT  ((size_t)16777216)  // [H][64][1024]
#define OFF_WKT  ((size_t)17825792)
#define OFF_WVT  ((size_t)18874368)
#define OFF_QB   ((size_t)19922944)  // [B][L][1024] bf16 inputs
#define OFF_KB   ((size_t)24117248)
#define OFF_VB   ((size_t)28311552)

#define ATTN_OUT_OFF ((size_t)4194304)   // f32 elems: ln_out region size

__device__ static inline bf16 f2bf(float f) { return (bf16)f; }

// A fragment: 16x32 bf16, row-major source; p = &A[m][k0]
__device__ static inline v16bf frag_a(const bf16* p, int hi) {
  union { v16bf v; v8bf h[2]; } u;
  u.h[0] = *(const v8bf*)(p + hi * 16);
  u.h[1] = *(const v8bf*)(p + hi * 16 + 8);
  return u.v;
}
// B fragment: 32x16 bf16; p = pointer to this lane's column (K-contiguous) + k0
__device__ static inline v16bf frag_b16(const bf16* p, int hi) {
  union { v16bf v; v8bf h[2]; } u;
  u.h[0] = *(const v8bf*)(p + hi * 16);
  u.h[1] = *(const v8bf*)(p + hi * 16 + 8);
  return u.v;
}
// B fragment from f32 (K-contiguous), converted on the fly
__device__ static inline v16bf frag_b_f32(const float* p, int hi) {
  v16bf r;
  const float* q = p + hi * 16;
#pragma unroll
  for (int e = 0; e < 16; ++e) r[e] = f2bf(q[e]);
  return r;
}
// A fragment from LDS f32 row; row = &S[m][k0]
__device__ static inline v16bf frag_a_lds(const float* row, int hi) {
  v16bf r;
#pragma unroll
  for (int e = 0; e < 8; ++e) r[e] = f2bf(row[hi * 16 + e]);
#pragma unroll
  for (int e = 0; e < 8; ++e) r[8 + e] = f2bf(row[hi * 16 + 8 + e]);
  return r;
}

__device__ static inline v8f wmma_bf16(v16bf a, v16bf b, v8f c) {
  return __builtin_amdgcn_wmma_f32_16x16x32_bf16(false, a, false, b, (short)0, c,
                                                 false, false);
}

// ---------------- prep: f32 -> bf16 convert ----------------
__global__ __launch_bounds__(256) void cvt_kernel(const float* __restrict__ src,
                                                  bf16* __restrict__ dst, int n) {
  int i = blockIdx.x * 256 + threadIdx.x;
  if (i < n) dst[i] = f2bf(src[i]);
}

// w: [16][1024][64] f32 -> wT: [16][64][1024] bf16
__global__ __launch_bounds__(256) void wtrans_kernel(const float* __restrict__ w,
                                                     bf16* __restrict__ wT) {
  int i = blockIdx.x * 256 + threadIdx.x;  // over 1,048,576
  int h = i >> 16;
  int rem = i & 65535;
  int nn = rem >> 10;
  int k = rem & 1023;
  wT[i] = f2bf(w[(size_t)h * 65536 + (size_t)k * 64 + nn]);
}

// ---------------- QKV per-head projections ----------------
// one block = (mat,h,b,ltile16); 4 waves, each a 16x16 N-tile, K=1024
__global__ __launch_bounds__(128) void qkv_proj_kernel(bf16* __restrict__ wsb) {
  int bx = blockIdx.x;
  int lt = bx & 127;
  int b = (bx >> 7) & 1;
  int h = (bx >> 8) & 15;
  int mat = bx >> 12;

  int tid = threadIdx.x;
  int lane = tid & 31;
  int wave = tid >> 5;
  int m = lane & 15;
  int hi = lane >> 4;
  int n = m;           // C/D column = lane & 15
  int l0 = lt * 16;
  int n0 = wave * 16;

  const bf16* xb = wsb + (mat == 0 ? OFF_QB : mat == 1 ? OFF_KB : OFF_VB);
  const bf16* wT = wsb + (mat == 0 ? OFF_WQT : mat == 1 ? OFF_WKT : OFF_WVT) +
                   (size_t)h * 65536;

  const bf16* arow = xb + ((size_t)(b * LL + l0 + m)) * DM;
  const bf16* bcol = wT + (size_t)(n0 + n) * DM;

  v8f c = {};
#pragma unroll 4
  for (int k0 = 0; k0 < DM; k0 += 32) {
    v16bf a = frag_a(arow + k0, hi);
    v16bf bb = frag_b16(bcol + k0, hi);
    c = wmma_bf16(a, bb, c);
  }

  if (mat < 2) {
    bf16* dst = wsb + (mat == 0 ? OFF_QS : OFF_KS) +
                ((size_t)(h * NB + b)) * LL * DK;
#pragma unroll
    for (int r = 0; r < 8; ++r) {
      int row = hi * 8 + r;
      dst[(size_t)(l0 + row) * DK + n0 + n] = f2bf(c[r]);
    }
  } else {
    bf16* dst = wsb + OFF_VST + ((size_t)(h * NB + b)) * DK * LL;
#pragma unroll
    for (int r = 0; r < 8; ++r) {
      int row = hi * 8 + r;
      dst[(size_t)(n0 + n) * LL + (l0 + row)] = f2bf(c[r]);
    }
  }
}

// ---------------- fused scores + softmax + PV ----------------
// one block = (h,b,16-row q tile); 4 waves; scores row-block kept in LDS
__global__ __launch_bounds__(128) void attn_kernel(bf16* __restrict__ wsb,
                                                   float* __restrict__ out) {
  __shared__ float S[16 * 2048];   // 128 KB (<= 320 KB/WGP on CDNA5)
  __shared__ float red[16][8];
  __shared__ float rowmax[16];
  __shared__ float rowsum[16];

  int bx = blockIdx.x;
  int qt = bx & 127;
  int b = (bx >> 7) & 1;
  int h = bx >> 8;
  int hb = h * NB + b;
  int q0 = qt * 16;

  int tid = threadIdx.x;
  int lane = tid & 31;
  int wave = tid >> 5;
  int m = lane & 15;
  int hi = lane >> 4;
  int n = m;

  const bf16* qs = wsb + OFF_QS + (size_t)hb * LL * DK;
  const bf16* ks = wsb + OFF_KS + (size_t)hb * LL * DK;
  const bf16* vsT = wsb + OFF_VST + (size_t)hb * DK * LL;

  // q fragments for this tile's 16 rows (K = 64 -> two 16x32 frags)
  const bf16* qrow = qs + (size_t)(q0 + m) * DK;
  v16bf aq0 = frag_a(qrow + 0, hi);
  v16bf aq1 = frag_a(qrow + 32, hi);

  // ---- scores: S[16 x 2048] = (q kT) / sqrt(d_model) ----
  for (int nt = wave; nt < 128; nt += 4) {
    int nb = nt * 16;
    const bf16* kcol = ks + (size_t)(nb + n) * DK;
    v16bf b0 = frag_b16(kcol + 0, hi);
    v16bf b1 = frag_b16(kcol + 32, hi);
    v8f c = {};
    c = wmma_bf16(aq0, b0, c);
    c = wmma_bf16(aq1, b1, c);
#pragma unroll
    for (int r = 0; r < 8; ++r)
      S[(hi * 8 + r) * 2048 + nb + n] = c[r] * 0.03125f;  // 1/sqrt(1024)
  }
  __syncthreads();

  // ---- softmax stats: 8 threads per row ----
  int tr = tid >> 3;
  int tc = tid & 7;
  float pmax = -3.4e38f;
  for (int j = tc * 256; j < tc * 256 + 256; ++j)
    pmax = fmaxf(pmax, S[tr * 2048 + j]);
  red[tr][tc] = pmax;
  __syncthreads();
  if (tc == 0) {
    float mx = red[tr][0];
#pragma unroll
    for (int i = 1; i < 8; ++i) mx = fmaxf(mx, red[tr][i]);
    rowmax[tr] = mx;
  }
  __syncthreads();
  float mx = rowmax[tr];
  float psum = 0.f;
  for (int j = tc * 256; j < tc * 256 + 256; ++j) {
    float e = __expf(S[tr * 2048 + j] - mx);
    S[tr * 2048 + j] = e;  // keep unnormalized exp in LDS
    psum += e;
  }
  red[tr][tc] = psum;
  __syncthreads();
  if (tc == 0) {
    float s = 0.f;
#pragma unroll
    for (int i = 0; i < 8; ++i) s += red[tr][i];
    rowsum[tr] = s;
  }
  __syncthreads();

  // ---- write normalized probs to d_out (coalesced) ----
  float* attns = out + ATTN_OUT_OFF + ((size_t)hb * LL + q0) * LL;
  for (int idx = tid; idx < 16 * 2048; idx += 128) {
    int r = idx >> 11;
    int j = idx & 2047;
    attns[(size_t)r * LL + j] = S[r * 2048 + j] * (1.0f / rowsum[r]);
  }

  // ---- PV: O(16x64) = P(16x2048) @ V(2048x64), per-wave 16x16 N-tile ----
  int nb = wave * 16;
  const bf16* vcol = vsT + (size_t)(nb + n) * LL;
  v8f o = {};
  for (int k0 = 0; k0 < 2048; k0 += 32) {
    v16bf a = frag_a_lds(&S[m * 2048 + k0], hi);
    v16bf bb = frag_b16(vcol + k0, hi);
    o = wmma_bf16(a, bb, o);
  }
  bf16* cat = wsb + OFF_CAT;
#pragma unroll
  for (int r = 0; r < 8; ++r) {
    int row = hi * 8 + r;
    float val = o[r] * (1.0f / rowsum[row]);
    cat[((size_t)(b * LL + q0 + row)) * DM + h * DK + nb + n] = f2bf(val);
  }
}

// ---------------- output projection + bias + residual ----------------
// one block = 32x64 tile of [4096 x 1024]; 8 waves (2 M x 4 N)
__global__ __launch_bounds__(256) void proj_kernel(const bf16* __restrict__ cat,
                                                   const float* __restrict__ projw,
                                                   const float* __restrict__ projb,
                                                   const float* __restrict__ qin,
                                                   float* __restrict__ zout) {
  int bx = blockIdx.x;
  int tm = bx & 127;
  int tn = bx >> 7;

  int tid = threadIdx.x;
  int lane = tid & 31;
  int wave = tid >> 5;
  int mrow = wave >> 2;
  int wcol = wave & 3;
  int m = lane & 15;
  int hi = lane >> 4;
  int n = m;

  int m0 = tm * 32 + mrow * 16;
  int n0 = tn * 64 + wcol * 16;

  const bf16* arow = cat + (size_t)(m0 + m) * DM;
  // out = cat @ proj_w.T  =>  B[k][col] = proj_w[col][k]  (K-contiguous rows)
  const float* bcol = projw + (size_t)(n0 + n) * DM;

  v8f c = {};
#pragma unroll 4
  for (int k0 = 0; k0 < DM; k0 += 32) {
    v16bf a = frag_a(arow + k0, hi);
    v16bf bb = frag_b_f32(bcol + k0, hi);
    c = wmma_bf16(a, bb, c);
  }

  int col = n0 + n;
  float bias = projb[col];
#pragma unroll
  for (int r = 0; r < 8; ++r) {
    int row = m0 + hi * 8 + r;
    float z = c[r] + bias + qin[(size_t)row * DM + col];  // residual
    zout[(size_t)row * DM + col] = z;
  }
}

// ---------------- LayerNorm (unbiased std, /(sigma+eps)) in-place ----------------
__global__ __launch_bounds__(256) void ln_kernel(float* __restrict__ z,
                                                 const float* __restrict__ ln_a,
                                                 const float* __restrict__ ln_b) {
  __shared__ float red[256];
  int row = blockIdx.x;
  int tid = threadIdx.x;
  float* base = z + (size_t)row * DM;

  float x[4];
#pragma unroll
  for (int i = 0; i < 4; ++i) x[i] = base[tid + i * 256];

  float s = x[0] + x[1] + x[2] + x[3];
  red[tid] = s;
  __syncthreads();
  for (int o = 128; o > 0; o >>= 1) {
    if (tid < o) red[tid] += red[tid + o];
    __syncthreads();
  }
  float mu = red[0] * (1.0f / DM);
  __syncthreads();

  float ss = 0.f;
#pragma unroll
  for (int i = 0; i < 4; ++i) {
    float d = x[i] - mu;
    ss += d * d;
  }
  red[tid] = ss;
  __syncthreads();
  for (int o = 128; o > 0; o >>= 1) {
    if (tid < o) red[tid] += red[tid + o];
    __syncthreads();
  }
  float sigma = sqrtf(red[0] / (float)(DM - 1));
  float sc = 1.0f / (sigma + 1e-3f);

#pragma unroll
  for (int i = 0; i < 4; ++i) {
    int j = tid + i * 256;
    base[j] = (x[i] - mu) * sc * ln_a[j] + ln_b[j];
  }
}

extern "C" void kernel_launch(void* const* d_in, const int* in_sizes, int n_in,
                              void* d_out, int out_size, void* d_ws, size_t ws_size,
                              hipStream_t stream) {
  (void)in_sizes; (void)n_in; (void)out_size; (void)ws_size;
  const float* q      = (const float*)d_in[0];
  const float* k      = (const float*)d_in[1];
  const float* v      = (const float*)d_in[2];
  const float* w_qs   = (const float*)d_in[3];
  const float* w_ks   = (const float*)d_in[4];
  const float* w_vs   = (const float*)d_in[5];
  const float* proj_w = (const float*)d_in[6];
  const float* proj_b = (const float*)d_in[7];
  const float* ln_a   = (const float*)d_in[8];
  const float* ln_b   = (const float*)d_in[9];
  float* out = (float*)d_out;
  bf16* wsb = (bf16*)d_ws;

  const int NELEM = NB * LL * DM;  // 4,194,304
  cvt_kernel<<<NELEM / 256, 256, 0, stream>>>(q, wsb + OFF_QB, NELEM);
  cvt_kernel<<<NELEM / 256, 256, 0, stream>>>(k, wsb + OFF_KB, NELEM);
  cvt_kernel<<<NELEM / 256, 256, 0, stream>>>(v, wsb + OFF_VB, NELEM);

  wtrans_kernel<<<1048576 / 256, 256, 0, stream>>>(w_qs, wsb + OFF_WQT);
  wtrans_kernel<<<1048576 / 256, 256, 0, stream>>>(w_ks, wsb + OFF_WKT);
  wtrans_kernel<<<1048576 / 256, 256, 0, stream>>>(w_vs, wsb + OFF_WVT);

  qkv_proj_kernel<<<3 * NH * NB * (LL / 16), 128, 0, stream>>>(wsb);
  attn_kernel<<<NH * NB * (LL / 16), 128, 0, stream>>>(wsb, out);
  proj_kernel<<<(NB * LL / 32) * (DM / 64), 256, 0, stream>>>(
      wsb + OFF_CAT, proj_w, proj_b, q, out);
  ln_kernel<<<NB * LL, 256, 0, stream>>>(out, ln_a, ln_b);
}
